// SoftBiasTransformer_8186207666973
// MI455X (gfx1250) — compile-verified, matrix operand order checked by reading, not run on hardware
//
#include <hip/hip_runtime.h>
#include <hip/hip_bf16.h>

typedef __bf16 bf16;
typedef __attribute__((ext_vector_type(16))) __bf16 v16bf;
typedef __attribute__((ext_vector_type(8)))  float   v8f;

static constexpr int Bz  = 256;
static constexpr int Nz  = 128;
static constexpr int Dz  = 384;
static constexpr int Hz  = 6;
static constexpr int DHz = 64;
static constexpr int Fz  = 1536;
static constexpr int Lz  = 6;
static constexpr int Mz  = Bz * Nz;     // 32768 tokens
static constexpr int QKV = 3 * Dz;      // 1152

#if defined(__AMDGCN__)
typedef int v4i_t __attribute__((vector_size(16)));
typedef __attribute__((address_space(1))) v4i_t as1_v4i;
typedef __attribute__((address_space(3))) v4i_t as3_v4i;
#endif

#if defined(__gfx1250__) && __has_builtin(__builtin_amdgcn_global_load_async_to_lds_b128)
#define USE_ASYNC_LDS 1
#else
#define USE_ASYNC_LDS 0
#endif

union FragBF { v16bf v; uint4 u[2]; };

static __device__ __forceinline__ v8f wmma_bf16(v16bf a, v16bf b, v8f c) {
  // v_wmma_f32_16x16x32_bf16  D = A(16x32) * B(32x16) + C
  return __builtin_amdgcn_wmma_f32_16x16x32_bf16(false, a, false, b, (short)0, c,
                                                 false, false);
}

// Load a 16x32 bf16 operand tile in the ISA 16-bit A/B layout:
// lane = row (mod 16), K-chunks {0..7,16..23} for lanes 0-15, {8..15,24..31} for 16-31.
static __device__ __forceinline__ v16bf load_frag(const bf16* tile, int stride, int lane) {
  const int r  = lane & 15;
  const int kh = (lane & 16) ? 8 : 0;
  const bf16* p = tile + (size_t)r * stride + kh;
  FragBF f;
  f.u[0] = *reinterpret_cast<const uint4*>(p);
  f.u[1] = *reinterpret_cast<const uint4*>(p + 16);
  return f.v;
}

// Async (or fallback) copy of one 16B chunk global -> LDS.
static __device__ __forceinline__ void stage_chunk(const bf16* gsrc, bf16* ldst) {
#if USE_ASYNC_LDS
  __builtin_amdgcn_global_load_async_to_lds_b128(
      (as1_v4i*)(void*)gsrc, (as3_v4i*)(void*)ldst, 0, 0);
#else
  *reinterpret_cast<uint4*>(ldst) = *reinterpret_cast<const uint4*>(gsrc);
#endif
}

static __device__ __forceinline__ void stage_fence() {
#if USE_ASYNC_LDS
#if __has_builtin(__builtin_amdgcn_s_wait_asynccnt)
  __builtin_amdgcn_s_wait_asynccnt(0);
#else
  asm volatile("s_wait_asynccnt 0" ::: "memory");
#endif
#endif
}

// ---------------------------------------------------------------------------
// bias[N,N] = sp_table[sp_matrix]
// ---------------------------------------------------------------------------
__global__ void build_bias_kernel(const int* __restrict__ spm,
                                  const float* __restrict__ spt,
                                  float* __restrict__ biasmat) {
  int i = blockIdx.x * blockDim.x + threadIdx.x;
  if (i < Nz * Nz) biasmat[i] = spt[spm[i]];
}

__global__ void cast_bf16_kernel(const float* __restrict__ src,
                                 bf16* __restrict__ dst, size_t n) {
  size_t i = (size_t)blockIdx.x * blockDim.x + threadIdx.x;
  if (i < n) dst[i] = (bf16)src[i];
}

__global__ void init_x_kernel(const float* __restrict__ xin,
                              float* __restrict__ x, bf16* __restrict__ xb) {
  size_t i = (size_t)blockIdx.x * blockDim.x + threadIdx.x;
  float v = xin[i];
  x[i]  = v;
  xb[i] = (bf16)v;
}

// ---------------------------------------------------------------------------
// GEMM: C[M,Nout] = A[M,K](bf16) * W[Nout,K]^T(bf16) + bias, optional relu.
// Block tile 128x128, 8 waves (4x2), each wave 32x64 = 2x4 WMMA tiles.
// K staged 32 wide in LDS with 40-element (80B) row pitch -> conflict-free.
// Staging uses GLOBAL_LOAD_ASYNC_TO_LDS_B128 when available.
// ---------------------------------------------------------------------------
__global__ void __launch_bounds__(256)
gemm_bf16_kernel(const bf16* __restrict__ A, const bf16* __restrict__ W,
                 const float* __restrict__ biasv, float* __restrict__ outF,
                 bf16* __restrict__ outB, int K, int Nout, int relu) {
  __shared__ bf16 Alds[128 * 40];
  __shared__ bf16 Wlds[128 * 40];
  const int tid  = threadIdx.x;
  const int lane = tid & 31;
  const int wid  = tid >> 5;
  const int m0 = blockIdx.x * 128;
  const int n0 = blockIdx.y * 128;
  const int mW = (wid & 3) * 32;
  const int nW = (wid >> 2) * 64;

  v8f acc[2][4];
#pragma unroll
  for (int i = 0; i < 2; ++i)
#pragma unroll
    for (int j = 0; j < 4; ++j)
#pragma unroll
      for (int e = 0; e < 8; ++e) acc[i][j][e] = 0.f;

  for (int k0 = 0; k0 < K; k0 += 32) {
    // stage A and W tiles: each 128x32 bf16 = 512 x 16B chunks
#pragma unroll
    for (int it = 0; it < 2; ++it) {
      int c   = tid + it * 256;
      int row = c >> 2;
      int ch  = (c & 3) * 8;
      stage_chunk(&A[(size_t)(m0 + row) * K + k0 + ch], &Alds[row * 40 + ch]);
      stage_chunk(&W[(size_t)(n0 + row) * K + k0 + ch], &Wlds[row * 40 + ch]);
#if !USE_ASYNC_LDS
      if (k0 + 32 < K)  // global_prefetch_b8 of next weight tile
        __builtin_prefetch(&W[(size_t)(n0 + row) * K + k0 + 32 + ch], 0, 1);
#endif
    }
    stage_fence();
    __syncthreads();

    v16bf a0 = load_frag(&Alds[(mW + 0)  * 40], 40, lane);
    v16bf a1 = load_frag(&Alds[(mW + 16) * 40], 40, lane);
#pragma unroll
    for (int j = 0; j < 4; ++j) {
      v16bf bj = load_frag(&Wlds[(nW + 16 * j) * 40], 40, lane);
      acc[0][j] = wmma_bf16(a0, bj, acc[0][j]);
      acc[1][j] = wmma_bf16(a1, bj, acc[1][j]);
    }
    __syncthreads();
  }

  const int rowHalf = (lane & 16) ? 8 : 0;
  const int col     = lane & 15;
#pragma unroll
  for (int i = 0; i < 2; ++i)
#pragma unroll
    for (int j = 0; j < 4; ++j) {
      const int n = n0 + nW + 16 * j + col;
      const float bn = biasv[n];
#pragma unroll
      for (int vg = 0; vg < 8; ++vg) {
        int m = m0 + mW + 16 * i + vg + rowHalf;
        float v = acc[i][j][vg] + bn;
        if (relu) v = fmaxf(v, 0.f);
        size_t idx = (size_t)m * Nout + n;
        if (outF) outF[idx] = v;
        if (outB) outB[idx] = (bf16)v;
      }
    }
}

// ---------------------------------------------------------------------------
// Attention for one (b,h): S = QK^T/8 + bias, softmax rows, O = P V.
// 8 waves; wave w owns query rows [16w,16w+16). All matmuls via WMMA.
// ---------------------------------------------------------------------------
__global__ void __launch_bounds__(256)
attn_kernel(const bf16* __restrict__ qkv, const float* __restrict__ biasmat,
            bf16* __restrict__ o) {
  __shared__ bf16 Ps[128 * 136];  // softmaxed probabilities, bf16
  __shared__ bf16 Vt[64 * 136];   // V transposed [d][k]
  const int tid  = threadIdx.x;
  const int lane = tid & 31;
  const int wid  = tid >> 5;
  const int b = blockIdx.x / Hz;
  const int h = blockIdx.x % Hz;
  const size_t tb = (size_t)b * Nz;  // token base

  // stage V^T (64 x 128)
  for (int idx = tid; idx < DHz * Nz; idx += 256) {
    int d  = idx >> 7;
    int kk = idx & 127;
    Vt[d * 136 + kk] = qkv[(tb + kk) * QKV + 2 * Dz + h * DHz + d];
  }
  __syncthreads();

  // ---- S = Q K^T ----
  v8f s[8];
#pragma unroll
  for (int t = 0; t < 8; ++t)
#pragma unroll
    for (int e = 0; e < 8; ++e) s[t][e] = 0.f;

  const bf16* Qbase = qkv + (tb + wid * 16) * QKV + h * DHz;
#pragma unroll
  for (int ks = 0; ks < 2; ++ks) {
    v16bf a = load_frag(Qbase + ks * 32, QKV, lane);
#pragma unroll
    for (int t = 0; t < 8; ++t) {
      const bf16* Kb = qkv + (tb + t * 16) * QKV + Dz + h * DHz + ks * 32;
      v16bf bf = load_frag(Kb, QKV, lane);
      s[t] = wmma_bf16(a, bf, s[t]);
    }
  }

  // ---- scale + pair bias + softmax (rows live in 16-lane halves) ----
  const int rowHalf = (lane & 16) ? 8 : 0;
  const int col     = lane & 15;
  float mrow[8];
#pragma unroll
  for (int j = 0; j < 8; ++j) mrow[j] = -1e30f;
#pragma unroll
  for (int t = 0; t < 8; ++t)
#pragma unroll
    for (int j = 0; j < 8; ++j) {
      int q = wid * 16 + j + rowHalf;
      int n = t * 16 + col;
      float v = s[t][j] * 0.125f + biasmat[q * Nz + n];
      s[t][j] = v;
      mrow[j] = fmaxf(mrow[j], v);
    }
#pragma unroll
  for (int j = 0; j < 8; ++j)
    for (int off = 8; off; off >>= 1)
      mrow[j] = fmaxf(mrow[j], __shfl_xor(mrow[j], off, 16));

  float rsum[8];
#pragma unroll
  for (int j = 0; j < 8; ++j) rsum[j] = 0.f;
#pragma unroll
  for (int t = 0; t < 8; ++t)
#pragma unroll
    for (int j = 0; j < 8; ++j) {
      float e = __expf(s[t][j] - mrow[j]);
      s[t][j] = e;
      rsum[j] += e;
    }
#pragma unroll
  for (int j = 0; j < 8; ++j) {
    for (int off = 8; off; off >>= 1) rsum[j] += __shfl_xor(rsum[j], off, 16);
    rsum[j] = 1.f / rsum[j];
  }
#pragma unroll
  for (int t = 0; t < 8; ++t)
#pragma unroll
    for (int j = 0; j < 8; ++j) {
      int q = wid * 16 + j + rowHalf;
      int n = t * 16 + col;
      Ps[q * 136 + n] = (bf16)(s[t][j] * rsum[j]);
    }
  __syncthreads();

  // ---- O = P V ----
  v8f oacc[4];
#pragma unroll
  for (int t = 0; t < 4; ++t)
#pragma unroll
    for (int e = 0; e < 8; ++e) oacc[t][e] = 0.f;

#pragma unroll
  for (int ks = 0; ks < 4; ++ks) {
    v16bf a = load_frag(&Ps[(wid * 16) * 136 + ks * 32], 136, lane);
#pragma unroll
    for (int tn = 0; tn < 4; ++tn) {
      v16bf bf = load_frag(&Vt[(tn * 16) * 136 + ks * 32], 136, lane);
      oacc[tn] = wmma_bf16(a, bf, oacc[tn]);
    }
  }
#pragma unroll
  for (int tn = 0; tn < 4; ++tn)
#pragma unroll
    for (int vg = 0; vg < 8; ++vg) {
      size_t q = tb + wid * 16 + vg + rowHalf;
      int d = tn * 16 + col;
      o[q * Dz + h * DHz + d] = (bf16)oacc[tn][vg];
    }
}

// ---------------------------------------------------------------------------
// x = LayerNorm(x + delta) * w + b ; also write bf16 shadow. 1 wave32 / token.
// ---------------------------------------------------------------------------
__global__ void __launch_bounds__(256)
residual_ln_kernel(float* __restrict__ x, const float* __restrict__ delta,
                   const float* __restrict__ w, const float* __restrict__ bb,
                   bf16* __restrict__ xb) {
  const int lane = threadIdx.x & 31;
  const size_t token = (size_t)blockIdx.x * 8 + (threadIdx.x >> 5);
  float v[12];
  float sum = 0.f;
#pragma unroll
  for (int i = 0; i < 12; ++i) {
    int d = lane + 32 * i;
    v[i] = x[token * Dz + d] + delta[token * Dz + d];
    sum += v[i];
  }
  for (int off = 16; off; off >>= 1) sum += __shfl_xor(sum, off, 32);
  float mu = sum * (1.f / Dz);
  float var = 0.f;
#pragma unroll
  for (int i = 0; i < 12; ++i) {
    float d0 = v[i] - mu;
    var += d0 * d0;
  }
  for (int off = 16; off; off >>= 1) var += __shfl_xor(var, off, 32);
  float rs = rsqrtf(var * (1.f / Dz) + 1e-5f);
#pragma unroll
  for (int i = 0; i < 12; ++i) {
    int d = lane + 32 * i;
    float y = (v[i] - mu) * rs * w[d] + bb[d];
    x[token * Dz + d]  = y;
    xb[token * Dz + d] = (bf16)y;
  }
}

// ---------------------------------------------------------------------------
extern "C" void kernel_launch(void* const* d_in, const int* in_sizes, int n_in,
                              void* d_out, int out_size, void* d_ws, size_t ws_size,
                              hipStream_t stream) {
  const float* x_in = (const float*)d_in[0];
  const int*   spm  = (const int*)d_in[1];
  const float* spt  = (const float*)d_in[2];
  const float* Wqkv = (const float*)d_in[3];
  const float* bqkv = (const float*)d_in[4];
  const float* Wo   = (const float*)d_in[5];
  const float* bo   = (const float*)d_in[6];
  const float* W1   = (const float*)d_in[7];
  const float* b1   = (const float*)d_in[8];
  const float* W2   = (const float*)d_in[9];
  const float* b2   = (const float*)d_in[10];
  const float* ln1w = (const float*)d_in[11];
  const float* ln1b = (const float*)d_in[12];
  const float* ln2w = (const float*)d_in[13];
  const float* ln2b = (const float*)d_in[14];
  float* xout = (float*)d_out;

  char* ws = (char*)d_ws;
  size_t off = 0;
  auto alloc = [&](size_t bytes) -> void* {
    void* p = ws + off;
    off += (bytes + 255) & ~(size_t)255;
    return p;
  };
  bf16*  WqkvB   = (bf16*)alloc((size_t)Lz * QKV * Dz * 2);
  bf16*  WoB     = (bf16*)alloc((size_t)Lz * Dz * Dz * 2);
  bf16*  W1B     = (bf16*)alloc((size_t)Lz * Fz * Dz * 2);
  bf16*  W2B     = (bf16*)alloc((size_t)Lz * Dz * Fz * 2);
  float* biasmat = (float*)alloc((size_t)Nz * Nz * 4);
  bf16*  xb      = (bf16*)alloc((size_t)Mz * Dz * 2);
  bf16*  qkvb    = (bf16*)alloc((size_t)Mz * QKV * 2);
  bf16*  ob      = (bf16*)alloc((size_t)Mz * Dz * 2);
  bf16*  hb      = (bf16*)alloc((size_t)Mz * Fz * 2);
  float* tmp     = (float*)alloc((size_t)Mz * Dz * 4);

  auto castN = [&](const float* s, bf16* d, size_t n) {
    cast_bf16_kernel<<<(unsigned)((n + 255) / 256), 256, 0, stream>>>(s, d, n);
  };

  build_bias_kernel<<<(Nz * Nz + 255) / 256, 256, 0, stream>>>(spm, spt, biasmat);
  castN(Wqkv, WqkvB, (size_t)Lz * QKV * Dz);
  castN(Wo,   WoB,   (size_t)Lz * Dz * Dz);
  castN(W1,   W1B,   (size_t)Lz * Fz * Dz);
  castN(W2,   W2B,   (size_t)Lz * Dz * Fz);
  init_x_kernel<<<(Mz * Dz) / 256, 256, 0, stream>>>(x_in, xout, xb);

  for (int l = 0; l < Lz; ++l) {
    // QKV projection -> qkvb (bf16)
    gemm_bf16_kernel<<<dim3(Mz / 128, QKV / 128), 256, 0, stream>>>(
        xb, WqkvB + (size_t)l * QKV * Dz, bqkv + (size_t)l * QKV,
        nullptr, qkvb, Dz, QKV, 0);
    // Attention -> ob (bf16)
    attn_kernel<<<Bz * Hz, 256, 0, stream>>>(qkvb, biasmat, ob);
    // Output projection -> tmp (f32)
    gemm_bf16_kernel<<<dim3(Mz / 128, Dz / 128), 256, 0, stream>>>(
        ob, WoB + (size_t)l * Dz * Dz, bo + (size_t)l * Dz,
        tmp, nullptr, Dz, Dz, 0);
    // x = LN(x + attn_out)
    residual_ln_kernel<<<Mz / 8, 256, 0, stream>>>(
        xout, tmp, ln1w + (size_t)l * Dz, ln1b + (size_t)l * Dz, xb);
    // FFN up + relu -> hb (bf16)
    gemm_bf16_kernel<<<dim3(Mz / 128, Fz / 128), 256, 0, stream>>>(
        xb, W1B + (size_t)l * Fz * Dz, b1 + (size_t)l * Fz,
        nullptr, hb, Dz, Fz, 1);
    // FFN down -> tmp (f32)
    gemm_bf16_kernel<<<dim3(Mz / 128, Dz / 128), 256, 0, stream>>>(
        hb, W2B + (size_t)l * Dz * Fz, b2 + (size_t)l * Dz,
        tmp, nullptr, Fz, Dz, 0);
    // x = LN(x + ffn_out)
    residual_ln_kernel<<<Mz / 8, 256, 0, stream>>>(
        xout, tmp, ln2w + (size_t)l * Dz, ln2b + (size_t)l * Dz, xb);
  }
}
